// Model_45552423142052
// MI455X (gfx1250) — compile-verified, matrix operand order checked by reading, not run on hardware
//
#include <hip/hip_runtime.h>
#include <hip/hip_bf16.h>
#include <stdint.h>

// ---------------------------------------------------------------------------
// GNN message passing on MI455X (gfx1250, wave32).
// Dense parts: v_wmma_f32_16x16x32_f16 (f16 in, f32 accumulate), epilogue
// writes f16 and/or f32 directly (no separate conversion passes).
// Edge aggregation: f16 gather (halved bandwidth) + f32 atomic scatter-add.
// ---------------------------------------------------------------------------

#define N_NODES 16384
#define N_EDGES 262144
#define DIM     300
#define DPAD    320           // padded feature dim (multiple of 32 for K-chunks)
#define N_GRAPHS 64

typedef __attribute__((ext_vector_type(16))) _Float16 v16h;
typedef __attribute__((ext_vector_type(8)))  float    v8f;
typedef __attribute__((ext_vector_type(2)))  _Float16 h2;
typedef __attribute__((ext_vector_type(4)))  _Float16 h4;

// --------------------------- conversion / padding ---------------------------

// dst[r*dcols + c] = (r<srows && c<scols) ? src[r*scols + c] : 0   (f32 -> f16)
__global__ void k_cvt_pad_f16(const float* __restrict__ src, _Float16* __restrict__ dst,
                              int srows, int scols, int drows, int dcols) {
    int i = blockIdx.x * blockDim.x + threadIdx.x;
    int total = drows * dcols;
    if (i >= total) return;
    int r = i / dcols, c = i % dcols;
    float v = (r < srows && c < scols) ? src[(size_t)r * scols + c] : 0.0f;
    dst[i] = (_Float16)v;
}

// Transposed weight convert: Wt[n*KPAD + k] = W[k*N + n], zero padded.
__global__ void k_cvt_w_t(const float* __restrict__ W, _Float16* __restrict__ Wt,
                          int K, int N, int KP, int NP) {
    int i = blockIdx.x * blockDim.x + threadIdx.x;
    int total = NP * KP;
    if (i >= total) return;
    int n = i / KP, k = i % KP;
    float v = (k < K && n < N) ? W[(size_t)k * N + n] : 0.0f;
    Wt[i] = (_Float16)v;
}

__global__ void k_pad_bias(const float* __restrict__ src, float* __restrict__ dst,
                           int n, int npad) {
    int i = blockIdx.x * blockDim.x + threadIdx.x;
    if (i >= npad) return;
    dst[i] = (i < n) ? src[i] : 0.0f;
}

__global__ void k_zero_f32(float* __restrict__ p, int n) {
    int i = blockIdx.x * blockDim.x + threadIdx.x;
    if (i < n) p[i] = 0.0f;
}

// ------------------------------- WMMA GEMM ----------------------------------
// act( A[M x DPAD](f16) * Bt^T + bias ) -> optional f32 Cf and/or f16 Ch.
// Bt is the K x N weight stored TRANSPOSED as [N x KPAD] f16 so that per-lane
// K-pairs are contiguous 32-bit loads for both operands.
// One wave computes one 16x16 tile; K looped in chunks of 32.
__global__ void k_gemm_bias_act(const _Float16* __restrict__ A,
                                const _Float16* __restrict__ Bt,
                                const float* __restrict__ bias,
                                float* __restrict__ Cf,        // may be null
                                _Float16* __restrict__ Ch,     // may be null
                                int M, int relu) {
    const int tiles_n = DPAD / 16;                       // 20
    int gtid = blockIdx.x * blockDim.x + threadIdx.x;
    int wave = gtid >> 5;
    int lane = threadIdx.x & 31;
    int tiles_m = M / 16;
    if (wave >= tiles_m * tiles_n) return;               // wave-uniform
    int tm = wave / tiles_n;
    int tn = wave % tiles_n;

    int idx = lane & 15;                                 // row (A) / col (B) in tile
    int grp = lane >> 4;                                 // lane half -> K half
    int arow = tm * 16 + idx;
    int bcol = tn * 16 + idx;

    const _Float16* __restrict__ arow_p = A  + (size_t)arow * DPAD;
    const _Float16* __restrict__ bcol_p = Bt + (size_t)bcol * DPAD;

    v8f acc = {};
    for (int k0 = 0; k0 < DPAD; k0 += 32) {
        v16h a, b;
#pragma unroll
        for (int j = 0; j < 8; ++j) {
            // ISA 16-bit A/B layout: lanes 0-15 K={0..7,16..23}, lanes 16-31 K={8..15,24..31}
            int kk = k0 + ((j & 3) * 2 + (j >> 2) * 16 + grp * 8);
            h2 av = *(const h2*)(arow_p + kk);
            h2 bv = *(const h2*)(bcol_p + kk);
            a[2 * j]     = av.x;
            a[2 * j + 1] = av.y;
            b[2 * j]     = bv.x;
            b[2 * j + 1] = bv.y;
        }
        acc = __builtin_amdgcn_wmma_f32_16x16x32_f16(
            /*neg_a=*/false, a, /*neg_b=*/false, b,
            /*c_mod=*/(short)0, acc, /*reuse_a=*/false, /*reuse_b=*/false);
    }

    float bv = bias[tn * 16 + idx];
    float res[8];
#pragma unroll
    for (int r = 0; r < 8; ++r) {
        float v = acc[r] + bv;
        if (relu) v = v > 0.0f ? v : 0.0f;
        res[r] = v;
    }
    int colofs = tn * 16 + idx;
    if (Cf) {
#pragma unroll
        for (int r = 0; r < 8; ++r) {
            int row = tm * 16 + r + 8 * grp;             // C/D VGPR layout
            Cf[(size_t)row * DPAD + colofs] = res[r];
        }
    }
    if (Ch) {
#pragma unroll
        for (int r = 0; r < 8; ++r) {
            int row = tm * 16 + r + 8 * grp;
            Ch[(size_t)row * DPAD + colofs] = (_Float16)res[r];
        }
    }
}

// --------------------------- edge scatter-add -------------------------------
// AGG[dst[e]] += T16[src[e]]  over real 300 cols (75 groups of 4 halves).
// Gather side is f16 (halved random-read bandwidth); accumulation is f32.
__global__ void k_edge_agg(const _Float16* __restrict__ T16, const int* __restrict__ src,
                           const int* __restrict__ dst, float* __restrict__ AGG) {
    int i = blockIdx.x * blockDim.x + threadIdx.x;
    const int total = N_EDGES * 75;
    if (i >= total) return;
    int e = i / 75;
    int c = (i % 75) * 4;
    int s = src[e], d = dst[e];
    h4 v = *(const h4*)(T16 + (size_t)s * DPAD + c);
    float* o = AGG + (size_t)d * DPAD + c;
    atomicAdd(o + 0, (float)v.x);
    atomicAdd(o + 1, (float)v.y);
    atomicAdd(o + 2, (float)v.z);
    atomicAdd(o + 3, (float)v.w);
}

// ------------------------------ readout -------------------------------------
// probs = softmax(h @ W_ro + b_ro); out[gid] += probs  (out pre-zeroed)
__global__ void k_readout(const float* __restrict__ H, const float* __restrict__ Wro,
                          const float* __restrict__ bro, const int* __restrict__ gid,
                          float* __restrict__ out) {
    int n = blockIdx.x * blockDim.x + threadIdx.x;
    if (n >= N_NODES) return;
    const float* h = H + (size_t)n * DPAD;
    float l0 = bro[0], l1 = bro[1];
    for (int k = 0; k < DIM; ++k) {
        float hv = h[k];
        l0 += hv * Wro[2 * k];
        l1 += hv * Wro[2 * k + 1];
    }
    float m  = fmaxf(l0, l1);
    float e0 = __expf(l0 - m);
    float e1 = __expf(l1 - m);
    float inv = 1.0f / (e0 + e1);
    int g = gid[n];
    atomicAdd(out + 2 * g,     e0 * inv);
    atomicAdd(out + 2 * g + 1, e1 * inv);
}

// ------------------------------- launcher -----------------------------------

extern "C" void kernel_launch(void* const* d_in, const int* in_sizes, int n_in,
                              void* d_out, int out_size, void* d_ws, size_t ws_size,
                              hipStream_t stream) {
    const float* node_feats = (const float*)d_in[0];
    const float* W_lift = (const float*)d_in[1];
    const float* b_lift = (const float*)d_in[2];
    const float* Wf[3] = { (const float*)d_in[3], (const float*)d_in[7],  (const float*)d_in[11] };
    const float* bf[3] = { (const float*)d_in[4], (const float*)d_in[8],  (const float*)d_in[12] };
    const float* Wo[3] = { (const float*)d_in[5], (const float*)d_in[9],  (const float*)d_in[13] };
    const float* bo[3] = { (const float*)d_in[6], (const float*)d_in[10], (const float*)d_in[14] };
    const float* W_ro = (const float*)d_in[15];
    const float* b_ro = (const float*)d_in[16];
    const int* src = (const int*)d_in[17];
    const int* dst = (const int*)d_in[18];
    const int* gid = (const int*)d_in[19];
    float* out = (float*)d_out;

    // ---- workspace layout (256B-aligned slices) ----
    char* ws = (char*)d_ws;
    size_t off = 0;
    _Float16* H16 = (_Float16*)(ws + off); off += (size_t)N_NODES * DPAD * 2;   // 10.5 MB
    _Float16* T16 = (_Float16*)(ws + off); off += (size_t)N_NODES * DPAD * 2;   // 10.5 MB
    float* F32A   = (float*)   (ws + off); off += (size_t)N_NODES * DPAD * 4;   // 21 MB (final h)
    float* AGG    = (float*)   (ws + off); off += (size_t)N_NODES * DPAD * 4;   // 21 MB
    _Float16* W16[7];
    for (int i = 0; i < 7; ++i) { W16[i] = (_Float16*)(ws + off); off += (size_t)DPAD * DPAD * 2; }
    float* Bp[7];
    for (int i = 0; i < 7; ++i) { Bp[i] = (float*)(ws + off); off += (size_t)DPAD * 4; }

    const int TB = 256;
    const int nH    = N_NODES * DPAD;                 // 5,242,880
    const int gH    = (nH + TB - 1) / TB;
    const int gW    = (DPAD * DPAD + TB - 1) / TB;
    const int gB    = (DPAD + TB - 1) / TB;
    const int tiles = (N_NODES / 16) * (DPAD / 16);   // 20480 waves
    const int gGemm = (tiles * 32 + TB - 1) / TB;     // 2560 blocks
    const int gEdge = (N_EDGES * 75 + TB - 1) / TB;   // 76800 blocks
    const int gNode = (N_NODES + TB - 1) / TB;

    // ---- one-time weight/bias conversion (transposed + padded f16) ----
    const float* Wsrc[7] = { W_lift, Wf[0], Wo[0], Wf[1], Wo[1], Wf[2], Wo[2] };
    const float* bsrc[7] = { b_lift, bf[0], bo[0], bf[1], bo[1], bf[2], bo[2] };
    for (int i = 0; i < 7; ++i) {
        k_cvt_w_t<<<gW, TB, 0, stream>>>(Wsrc[i], W16[i], DIM, DIM, DPAD, DPAD);
        k_pad_bias<<<gB, TB, 0, stream>>>(bsrc[i], Bp[i], DIM, DPAD);
    }

    // ---- lift: h = X @ W_lift + b  (f16 out only) ----
    k_cvt_pad_f16<<<gH, TB, 0, stream>>>(node_feats, T16, N_NODES, DIM, N_NODES, DPAD);
    k_gemm_bias_act<<<gGemm, TB, 0, stream>>>(T16, W16[0], Bp[0],
                                              (float*)nullptr, H16, N_NODES, /*relu=*/0);

    // ---- 3 message-passing layers ----
    for (int L = 0; L < 3; ++L) {
        _Float16* Wf16 = W16[1 + 2 * L];  float* bfp = Bp[1 + 2 * L];
        _Float16* Wo16 = W16[2 + 2 * L];  float* bop = Bp[2 + 2 * L];
        int last = (L == 2);

        // t = relu(h @ Wf + bf)  (per-node; gather commutes with the dense op)
        k_gemm_bias_act<<<gGemm, TB, 0, stream>>>(H16, Wf16, bfp,
                                                  (float*)nullptr, T16, N_NODES, /*relu=*/1);
        // agg[dst] += t[src]   (f16 gather, f32 atomic accumulate)
        k_zero_f32<<<gH, TB, 0, stream>>>(AGG, nH);
        k_edge_agg<<<gEdge, TB, 0, stream>>>(T16, src, dst, AGG);
        // h = relu(agg @ Wo + bo)
        k_cvt_pad_f16<<<gH, TB, 0, stream>>>(AGG, T16, N_NODES, DPAD, N_NODES, DPAD);
        k_gemm_bias_act<<<gGemm, TB, 0, stream>>>(T16, Wo16, bop,
                                                  last ? F32A : (float*)nullptr,
                                                  last ? (_Float16*)nullptr : H16,
                                                  N_NODES, /*relu=*/1);
    }

    // ---- readout: softmax + segment-sum into [64 x 2] ----
    k_zero_f32<<<1, 128, 0, stream>>>(out, N_GRAPHS * 2);
    k_readout<<<gNode, TB, 0, stream>>>(F32A, W_ro, b_ro, gid, out);
}
